// GATLayer_14834817040882
// MI455X (gfx1250) — matrix operand order repaired and checked by reading, not compiled, not run on hardware
//
#include <hip/hip_runtime.h>
#include <hip/hip_bf16.h>

typedef _Float16 v16h __attribute__((ext_vector_type(16)));
typedef _Float16 h8   __attribute__((ext_vector_type(8)));
typedef float    v8f  __attribute__((ext_vector_type(8)));
typedef float    f8   __attribute__((ext_vector_type(8)));

#define N_    4096
#define IN_   256
#define H_    8
#define D_    64
#define HD_   512   // H_*D_
#define SLOPE 0.2f

// ---------------------------------------------------------------- converts
__global__ void cvt_f32_f16_kernel(const float* __restrict__ src,
                                   _Float16* __restrict__ dst, int n) {
    int i = blockIdx.x * blockDim.x + threadIdx.x;
    if (i < n) dst[i] = (_Float16)src[i];
}

// ---------------------------------------------------------------- GEMM1: Wx = x @ W^T
// One wave computes a 16x16 tile of Wx (M=4096, N=512, K=256).
// A = xh row-major [4096][256]; B(k,n) = W[n][k] -> wh row-major [512][256].
// Outputs: Wx f32 [4096][512] and WxT f16 [512][4096] (per-head transposed for phase 2).
__global__ void gemm1_kernel(const _Float16* __restrict__ xh,
                             const _Float16* __restrict__ wh,
                             float* __restrict__ Wx,
                             _Float16* __restrict__ WxT) {
    int wave = blockIdx.x * 8 + (threadIdx.x >> 5);   // 8192 waves
    int lane = threadIdx.x & 31;
    int nTile = wave & 31;          // 512/16 = 32
    int mTile = wave >> 5;          // 4096/16 = 256
    int mBase = mTile * 16, nBase = nTile * 16;
    int lm = lane & 15, lg = lane >> 4;

    const _Float16* aRow = xh + (size_t)(mBase + lm) * IN_;
    const _Float16* bRow = wh + (size_t)(nBase + lm) * IN_;

    v8f c = {};
    #pragma unroll
    for (int k = 0; k < IN_; k += 32) {
        // A fragment: lane row m=lm; halves 0..7 -> k+lg*8+q, halves 8..15 -> k+16+lg*8+q
        h8 a0 = *(const h8*)(aRow + k + lg * 8);
        h8 a1 = *(const h8*)(aRow + k + lg * 8 + 16);
        v16h a;
        #pragma unroll
        for (int q = 0; q < 8; ++q) { a[q] = a0[q]; a[q + 8] = a1[q]; }
        // B fragment: lane col n=lm; halves 0..7 -> k+lg*8+q, halves 8..15 -> k+16+lg*8+q
        h8 b0 = *(const h8*)(bRow + k + lg * 8);
        h8 b1 = *(const h8*)(bRow + k + lg * 8 + 16);
        v16h b;
        #pragma unroll
        for (int q = 0; q < 8; ++q) { b[q] = b0[q]; b[q + 8] = b1[q]; }
        c = __builtin_amdgcn_wmma_f32_16x16x32_f16(false, a, false, b,
                                                   (short)0, c, false, false);
    }
    // C/D layout: lane = n + 16*(m/8); VGPR v holds m = v + 8*lg, n = lm
    #pragma unroll
    for (int v = 0; v < 8; ++v) {
        int m = mBase + v + 8 * lg;
        int n = nBase + lm;
        Wx [(size_t)m * HD_ + n] = c[v];
        WxT[(size_t)n * N_  + m] = (_Float16)c[v];
    }
}

// ---------------------------------------------------------------- s, t, E1, E2
__global__ void st_kernel(const float* __restrict__ Wx, const float* __restrict__ attn,
                          float* __restrict__ sArr, float* __restrict__ tHead,
                          float* __restrict__ E1, float* __restrict__ E2) {
    int idx = blockIdx.x * blockDim.x + threadIdx.x;
    if (idx >= N_ * H_) return;
    int i = idx / H_, h = idx % H_;
    const float* row = Wx + (size_t)i * HD_ + h * D_;
    const float* as  = attn + h * 2 * D_;
    const float* at  = as + D_;
    float s = 0.f, t = 0.f;
    #pragma unroll 8
    for (int d = 0; d < D_; ++d) { s += row[d] * as[d]; t += row[d] * at[d]; }
    sArr [(size_t)i * H_ + h] = s;
    tHead[(size_t)h * N_ + i] = t;
    E1   [(size_t)h * N_ + i] = expf(t);
    E2   [(size_t)h * N_ + i] = expf(SLOPE * t);
}

// ---------------------------------------------------------------- per-head max of t
__global__ void tmax_kernel(const float* __restrict__ tHead, float* __restrict__ tmax) {
    __shared__ float red[256];
    int h = blockIdx.x;
    float m = -1e30f;
    for (int i = threadIdx.x; i < N_; i += 256) m = fmaxf(m, tHead[(size_t)h * N_ + i]);
    red[threadIdx.x] = m;
    __syncthreads();
    for (int s = 128; s > 0; s >>= 1) {
        if (threadIdx.x < s) red[threadIdx.x] = fmaxf(red[threadIdx.x], red[threadIdx.x + s]);
        __syncthreads();
    }
    if (threadIdx.x == 0) tmax[h] = red[0];
}

// ---------------------------------------------------------------- fused softmax-aggregate
// One wave handles (16-row i-tile, head h): out[i,h,:] = (1/Z_i) sum_j p(i,j) Wx[j,h,:]
// p = exp(leaky_relu(s_i+t_j) - m_i) = max(c1*E1_j, c2*E2_j); streamed over j in steps of 32.
__global__ void aggr_kernel(const _Float16* __restrict__ WxT,
                            const float* __restrict__ sArr,
                            const float* __restrict__ tmax,
                            const float* __restrict__ E1,
                            const float* __restrict__ E2,
                            float* __restrict__ out) {
    int wave = blockIdx.x * 8 + (threadIdx.x >> 5);   // 2048 waves
    int lane = threadIdx.x & 31;
    int h = wave & (H_ - 1);
    int iTile = wave >> 3;
    int lm = lane & 15, lg = lane >> 4;

    float s  = sArr[(size_t)(iTile * 16 + lm) * H_ + h];
    float tm = tmax[h];
    float sm = s + tm;
    float m  = sm >= 0.f ? sm : SLOPE * sm;           // m_i = lr(s_i + tmax) >= all scores
    float c1 = expf(s - m);
    float c2 = expf(SLOPE * s - m);

    const float* e1 = E1 + (size_t)h * N_ + lg * 16;
    const float* e2 = E2 + (size_t)h * N_ + lg * 16;
    // B fragments: B(k,n) = WxT[(h*64 + db*16 + n)*4096 + jB + k]; lane col n=lm, k = lg*16+hi
    const _Float16* bp0 = WxT + (size_t)(h * D_ +  0 + lm) * N_ + lg * 16;
    const _Float16* bp1 = WxT + (size_t)(h * D_ + 16 + lm) * N_ + lg * 16;
    const _Float16* bp2 = WxT + (size_t)(h * D_ + 32 + lm) * N_ + lg * 16;
    const _Float16* bp3 = WxT + (size_t)(h * D_ + 48 + lm) * N_ + lg * 16;

    v8f acc0 = {}, acc1 = {}, acc2 = {}, acc3 = {};
    float z = 0.f;

    for (int jB = 0; jB < N_; jB += 32) {
        f8 x1a = *(const f8*)(e1 + jB);        // k = lg*16 + 0..7
        f8 x1b = *(const f8*)(e1 + jB + 8);    // k = lg*16 + 8 + 0..7
        f8 x2a = *(const f8*)(e2 + jB);
        f8 x2b = *(const f8*)(e2 + jB + 8);
        v16h a;
        float zl = 0.f;
        #pragma unroll
        for (int q = 0; q < 8; ++q) {
            float pa = fmaxf(c1 * x1a[q], c2 * x2a[q]);
            float pb = fmaxf(c1 * x1b[q], c2 * x2b[q]);
            zl += pa + pb;
            a[q]     = (_Float16)pa;
            a[q + 8] = (_Float16)pb;
        }
        z += zl;
        v16h b0 = *(const v16h*)(bp0 + jB);
        v16h b1 = *(const v16h*)(bp1 + jB);
        v16h b2 = *(const v16h*)(bp2 + jB);
        v16h b3 = *(const v16h*)(bp3 + jB);
        acc0 = __builtin_amdgcn_wmma_f32_16x16x32_f16(false, a, false, b0, (short)0, acc0, false, false);
        acc1 = __builtin_amdgcn_wmma_f32_16x16x32_f16(false, a, false, b1, (short)0, acc1, false, false);
        acc2 = __builtin_amdgcn_wmma_f32_16x16x32_f16(false, a, false, b2, (short)0, acc2, false, false);
        acc3 = __builtin_amdgcn_wmma_f32_16x16x32_f16(false, a, false, b3, (short)0, acc3, false, false);
    }

    // Row normalizer: lane L has partial for row lm over half the k's; pair with lane^16.
    float zFull = z + __shfl_xor(z, 16, 32);   // now lane L holds Z for row lm
    #pragma unroll
    for (int v = 0; v < 8; ++v) {
        int mrow = v + 8 * lg;                 // D layout: VGPR v -> m = v + 8*(lane/16)
        float Zv = __shfl(zFull, mrow, 32);    // row mrow's Z lives in lane mrow
        float inv = 1.0f / Zv;
        size_t o = (size_t)(iTile * 16 + mrow) * HD_ + h * D_ + lm;
        out[o +  0] = acc0[v] * inv;
        out[o + 16] = acc1[v] * inv;
        out[o + 32] = acc2[v] * inv;
        out[o + 48] = acc3[v] * inv;
    }
}

// ---------------------------------------------------------------- launch
extern "C" void kernel_launch(void* const* d_in, const int* in_sizes, int n_in,
                              void* d_out, int out_size, void* d_ws, size_t ws_size,
                              hipStream_t stream) {
    const float* x    = (const float*)d_in[0];   // [4096][256]
    const float* W    = (const float*)d_in[1];   // [512][256]
    const float* attn = (const float*)d_in[2];   // [1][8][128]
    float* out = (float*)d_out;                  // [4096][512]

    char* p = (char*)d_ws;
    auto carve = [&](size_t bytes) {
        void* r = (void*)p;
        p += (bytes + 255) & ~(size_t)255;
        return r;
    };
    _Float16* xh    = (_Float16*)carve((size_t)N_ * IN_ * 2);
    _Float16* wh    = (_Float16*)carve((size_t)HD_ * IN_ * 2);
    float*    Wx    = (float*)   carve((size_t)N_ * HD_ * 4);
    _Float16* WxT   = (_Float16*)carve((size_t)HD_ * N_ * 2);
    float*    sArr  = (float*)   carve((size_t)N_ * H_ * 4);
    float*    tHead = (float*)   carve((size_t)N_ * H_ * 4);
    float*    E1    = (float*)   carve((size_t)H_ * N_ * 4);
    float*    E2    = (float*)   carve((size_t)H_ * N_ * 4);
    float*    tmx   = (float*)   carve(H_ * 4);

    cvt_f32_f16_kernel<<<(N_ * IN_ + 255) / 256, 256, 0, stream>>>(x, xh, N_ * IN_);
    cvt_f32_f16_kernel<<<(HD_ * IN_ + 255) / 256, 256, 0, stream>>>(W, wh, HD_ * IN_);

    // 8192 wave-tiles, 8 waves per 256-thread block
    gemm1_kernel<<<1024, 256, 0, stream>>>(xh, wh, Wx, WxT);

    st_kernel<<<(N_ * H_ + 255) / 256, 256, 0, stream>>>(Wx, attn, sArr, tHead, E1, E2);
    tmax_kernel<<<H_, 256, 0, stream>>>(tHead, tmx);

    // 2048 wave-tasks (256 i-tiles x 8 heads), 8 waves per block
    aggr_kernel<<<256, 256, 0, stream>>>(WxT, sArr, tmx, E1, E2, out);
}
